// MyFunction_52699248722437
// MI455X (gfx1250) — matrix operand order repaired and checked by reading, not compiled, not run on hardware
//
#include <hip/hip_runtime.h>
#include <hip/hip_bf16.h>

// ---------------------------------------------------------------------------
// Problem constants (match reference)
// ---------------------------------------------------------------------------
#define BB     1024            // batch
#define CIN    1024
#define COUT   2048
#define KK     49              // 7x7 taps
#define KDIM   (CIN * KK)      // 50176 : GEMM K for offset + deform conv
#define NOFF   128             // padded offset-channel count (98 -> 128)

typedef __bf16 bf16_t;
typedef bf16_t bf16x4  __attribute__((ext_vector_type(4)));
typedef bf16_t bf16x8  __attribute__((ext_vector_type(8)));
typedef bf16_t bf16x16 __attribute__((ext_vector_type(16)));
typedef float  f32x8   __attribute__((ext_vector_type(8)));

// ---------------------------------------------------------------------------
// fp32 -> bf16 converters
// ---------------------------------------------------------------------------
__global__ __launch_bounds__(256) void cvt4_f32_bf16(
    const float* __restrict__ in, bf16_t* __restrict__ out, long n4) {
  long i = (long)blockIdx.x * blockDim.x + threadIdx.x;
  if (i < n4) {
    float4 v = reinterpret_cast<const float4*>(in)[i];
    bf16x4 o = { (bf16_t)v.x, (bf16_t)v.y, (bf16_t)v.z, (bf16_t)v.w };
    reinterpret_cast<bf16x4*>(out)[i] = o;
  }
}

// offset conv weight: [98, KDIM] fp32 -> [128, KDIM] bf16, rows 98..127 zeroed
__global__ __launch_bounds__(256) void cvt_pad_offw(
    const float* __restrict__ in, bf16_t* __restrict__ out, long n) {
  long i = (long)blockIdx.x * blockDim.x + threadIdx.x;
  if (i < n) {
    long row = i / KDIM;
    long k   = i - row * KDIM;
    float v  = (row < 2 * KK) ? in[row * (long)KDIM + k] : 0.0f;
    out[i] = (bf16_t)v;
  }
}

// ---------------------------------------------------------------------------
// im2col with bilinear deformable sampling: cols[b, c*49 + p] (bf16)
// offsets buffer is fp32 [B, NOFF]; channel (2p)=dy, (2p+1)=dx.
// Output spatial is 1x1 (H=W=K, stride1, pad0) so base position is (0,0).
// ---------------------------------------------------------------------------
__global__ __launch_bounds__(256) void build_cols(
    const float* __restrict__ x, const float* __restrict__ offs,
    bf16_t* __restrict__ cols) {
  __shared__ float4 s_w[KK];   // 4 bilinear weights per tap (0 if invalid)
  __shared__ int4   s_i[KK];   // 4 clamped spatial indices per tap
  const int b = blockIdx.x;

  if (threadIdx.x < KK) {
    int p = threadIdx.x;
    float dy = offs[(long)b * NOFF + 2 * p];
    float dx = offs[(long)b * NOFF + 2 * p + 1];
    float py = dy + (float)(p / 7);
    float px = dx + (float)(p % 7);
    float y0f = floorf(py), x0f = floorf(px);
    float fy = py - y0f,    fx = px - x0f;
    int y0 = (int)y0f, x0 = (int)x0f;
    int y1 = y0 + 1,   x1 = x0 + 1;
    bool vy0 = (y0 >= 0) & (y0 < 7), vy1 = (y1 >= 0) & (y1 < 7);
    bool vx0 = (x0 >= 0) & (x0 < 7), vx1 = (x1 >= 0) & (x1 < 7);
    int cy0 = min(max(y0, 0), 6), cy1 = min(max(y1, 0), 6);
    int cx0 = min(max(x0, 0), 6), cx1 = min(max(x1, 0), 6);
    float4 w;
    w.x = (vy0 & vx0) ? (1.f - fy) * (1.f - fx) : 0.f;
    w.y = (vy0 & vx1) ? (1.f - fy) * fx         : 0.f;
    w.z = (vy1 & vx0) ? fy * (1.f - fx)         : 0.f;
    w.w = (vy1 & vx1) ? fy * fx                 : 0.f;
    s_w[p] = w;
    s_i[p] = make_int4(cy0 * 7 + cx0, cy0 * 7 + cx1,
                       cy1 * 7 + cx0, cy1 * 7 + cx1);
  }
  __syncthreads();

  const float* xb = x + (long)b * CIN * KK;
  bf16_t* cb = cols + (long)b * KDIM;
  for (int k = threadIdx.x; k < KDIM; k += 256) {
    int c = k / KK;
    int p = k - c * KK;
    float4 w = s_w[p];
    int4   ii = s_i[p];
    const float* xp = xb + (long)c * KK;
    float v = xp[ii.x] * w.x + xp[ii.y] * w.y +
              xp[ii.z] * w.z + xp[ii.w] * w.w;
    cb[k] = (bf16_t)v;
  }
}

// ---------------------------------------------------------------------------
// bf16 WMMA fragment loaders, per ISA 7.12.2 wave32 layouts.
// A (16x32, 16-bit): lane L -> row (L&15); chunks K+[hi*8 .. +7] and
//                    K+[16+hi*8 .. +7]  (hi = L>>4)
// B (32x16, 16-bit): lane L -> col (L&15); chunks K+[hi*8 .. +7] and
//                    K+[16+hi*8 .. +7]  (hi = L>>4)
// Both A and the weight matrices are stored row-major [rows, K], so every
// chunk is a contiguous 16-byte global_load_b128.
// ---------------------------------------------------------------------------
__device__ __forceinline__ bf16x16 load_fragA(
    const bf16_t* __restrict__ A, long ld, long row0, long k, int lane) {
  long r = row0 + (lane & 15);
  int hi = lane >> 4;
  const bf16_t* p = A + r * ld + k + hi * 8;
  bf16x8 lo  = *reinterpret_cast<const bf16x8*>(p);
  bf16x8 hi8 = *reinterpret_cast<const bf16x8*>(p + 16);
  return __builtin_shufflevector(lo, hi8, 0,1,2,3,4,5,6,7,8,9,10,11,12,13,14,15);
}

__device__ __forceinline__ bf16x16 load_fragB(
    const bf16_t* __restrict__ Bw, long ld, long n0, long k, int lane) {
  long n = n0 + (lane & 15);
  int hi = lane >> 4;
  const bf16_t* p = Bw + n * ld + k + hi * 8;
  bf16x8 lo  = *reinterpret_cast<const bf16x8*>(p);
  bf16x8 hi8 = *reinterpret_cast<const bf16x8*>(p + 16);
  return __builtin_shufflevector(lo, hi8, 0,1,2,3,4,5,6,7,8,9,10,11,12,13,14,15);
}

#define WMMA_BF16(a, b, c) \
  __builtin_amdgcn_wmma_f32_16x16x32_bf16(false, (a), false, (b), (short)0, (c), false, false)

// ---------------------------------------------------------------------------
// Generic bf16 GEMM:  out[m,n] = sum_k A[m,k] * Bw[n,k]   (fp32 accum)
// Block = 128 threads = 4 waves (2 M x 2 N). Wave tile = 32x64 (2x4 WMMA).
// Block tile = 64 x 128. M%64==0, N%128==0, K%32==0 guaranteed by caller,
// so EXEC is all-ones everywhere (WMMA requirement).
// MODE 0: outF = acc                  (offset conv, no bias)
// MODE 1: outB = bf16(acc + bias[n])  (deform conv -> feeds linear GEMM)
// MODE 2: outF = softshrink(acc + bias[n], 0.5)   (final output)
// ---------------------------------------------------------------------------
template <int MODE>
__global__ __launch_bounds__(128) void gemm_bf16_wmma(
    const bf16_t* __restrict__ A, const bf16_t* __restrict__ Bw,
    const float* __restrict__ bias, float* __restrict__ outF,
    bf16_t* __restrict__ outB, int N, long K) {
  const int lane  = threadIdx.x & 31;
  const int wave  = threadIdx.x >> 5;
  const int waveM = wave & 1;
  const int waveN = wave >> 1;
  const long mbase = (long)blockIdx.x * 64 + waveM * 32;
  const long nbase = (long)blockIdx.y * 128 + waveN * 64;

  f32x8 acc00 = {}, acc01 = {}, acc02 = {}, acc03 = {};
  f32x8 acc10 = {}, acc11 = {}, acc12 = {}, acc13 = {};

  for (long k = 0; k < K; k += 32) {
    bf16x16 a0 = load_fragA(A, K, mbase,      k, lane);
    bf16x16 a1 = load_fragA(A, K, mbase + 16, k, lane);
    bf16x16 b0 = load_fragB(Bw, K, nbase,      k, lane);
    bf16x16 b1 = load_fragB(Bw, K, nbase + 16, k, lane);
    bf16x16 b2 = load_fragB(Bw, K, nbase + 32, k, lane);
    bf16x16 b3 = load_fragB(Bw, K, nbase + 48, k, lane);
    acc00 = WMMA_BF16(a0, b0, acc00);
    acc01 = WMMA_BF16(a0, b1, acc01);
    acc02 = WMMA_BF16(a0, b2, acc02);
    acc03 = WMMA_BF16(a0, b3, acc03);
    acc10 = WMMA_BF16(a1, b0, acc10);
    acc11 = WMMA_BF16(a1, b1, acc11);
    acc12 = WMMA_BF16(a1, b2, acc12);
    acc13 = WMMA_BF16(a1, b3, acc13);
  }

  // C/D layout: VGPR v, lanes 0-15 -> M=v, lanes 16-31 -> M=v+8; N = lane&15.
  const int col = lane & 15;
  const int hi  = lane >> 4;

#define EPILOGUE_TILE(ACC, MT, NT)                                          \
  {                                                                         \
    long n = nbase + (NT) * 16 + col;                                       \
    float bv = 0.0f;                                                        \
    if constexpr (MODE != 0) bv = bias[n];                                  \
    _Pragma("unroll")                                                       \
    for (int v = 0; v < 8; ++v) {                                           \
      long m = mbase + (MT) * 16 + hi * 8 + v;                              \
      float val = (ACC)[v];                                                 \
      if constexpr (MODE == 0) {                                            \
        outF[m * (long)N + n] = val;                                        \
      } else if constexpr (MODE == 1) {                                     \
        outB[m * (long)N + n] = (bf16_t)(val + bv);                         \
      } else {                                                              \
        float t = val + bv;                                                 \
        float y = (t > 0.5f) ? (t - 0.5f)                                   \
                : ((t < -0.5f) ? (t + 0.5f) : 0.0f);                        \
        outF[m * (long)N + n] = y;                                          \
      }                                                                     \
    }                                                                       \
  }

  EPILOGUE_TILE(acc00, 0, 0)
  EPILOGUE_TILE(acc01, 0, 1)
  EPILOGUE_TILE(acc02, 0, 2)
  EPILOGUE_TILE(acc03, 0, 3)
  EPILOGUE_TILE(acc10, 1, 0)
  EPILOGUE_TILE(acc11, 1, 1)
  EPILOGUE_TILE(acc12, 1, 2)
  EPILOGUE_TILE(acc13, 1, 3)
#undef EPILOGUE_TILE
}

// ---------------------------------------------------------------------------
// Host-side launch
// ---------------------------------------------------------------------------
extern "C" void kernel_launch(void* const* d_in, const int* in_sizes, int n_in,
                              void* d_out, int out_size, void* d_ws, size_t ws_size,
                              hipStream_t stream) {
  (void)in_sizes; (void)n_in; (void)out_size; (void)ws_size;
  const float* x        = (const float*)d_in[0];   // [B, CIN, 7, 7]
  const float* offset_w = (const float*)d_in[1];   // [98, CIN, 7, 7]
  const float* deform_w = (const float*)d_in[2];   // [COUT, CIN, 7, 7]
  const float* deform_b = (const float*)d_in[3];   // [COUT]
  const float* lin_w    = (const float*)d_in[4];   // [COUT, COUT]
  const float* lin_b    = (const float*)d_in[5];   // [COUT]
  float* out = (float*)d_out;                      // [B, COUT] fp32

  // workspace carve-up (256B aligned)
  char* ws = (char*)d_ws;
  size_t cur = 0;
  auto take = [&](size_t bytes) -> char* {
    char* p = ws + cur;
    cur += (bytes + 255) & ~(size_t)255;
    return p;
  };
  bf16_t* xcols   = (bf16_t*)take((size_t)BB * KDIM * 2);     // xbf, later cols
  bf16_t* offwbf  = (bf16_t*)take((size_t)NOFF * KDIM * 2);
  bf16_t* wbf     = (bf16_t*)take((size_t)COUT * KDIM * 2);
  bf16_t* linwbf  = (bf16_t*)take((size_t)COUT * COUT * 2);
  float*  offsets = (float*) take((size_t)BB * NOFF * 4);
  bf16_t* out1bf  = (bf16_t*)take((size_t)BB * COUT * 2);

  // 1) fp32 -> bf16 conversions
  {
    long n4 = (long)BB * KDIM / 4;
    cvt4_f32_bf16<<<(unsigned)((n4 + 255) / 256), 256, 0, stream>>>(x, xcols, n4);
  }
  {
    long n = (long)NOFF * KDIM;
    cvt_pad_offw<<<(unsigned)((n + 255) / 256), 256, 0, stream>>>(offset_w, offwbf, n);
  }
  {
    long n4 = (long)COUT * KDIM / 4;
    cvt4_f32_bf16<<<(unsigned)((n4 + 255) / 256), 256, 0, stream>>>(deform_w, wbf, n4);
  }
  {
    long n4 = (long)COUT * COUT / 4;
    cvt4_f32_bf16<<<(unsigned)((n4 + 255) / 256), 256, 0, stream>>>(lin_w, linwbf, n4);
  }

  // 2) offset conv as WMMA GEMM: [1024, 128(pad)] = xbf[1024,50176] @ offwbf^T
  gemm_bf16_wmma<0><<<dim3(BB / 64, NOFF / 128), 128, 0, stream>>>(
      xcols, offwbf, nullptr, offsets, nullptr, NOFF, (long)KDIM);

  // 3) deformable im2col (overwrites xbf buffer with cols; stream-ordered)
  build_cols<<<BB, 256, 0, stream>>>(x, offsets, xcols);

  // 4) deform conv GEMM + bias -> bf16: [1024, 2048]
  gemm_bf16_wmma<1><<<dim3(BB / 64, COUT / 128), 128, 0, stream>>>(
      xcols, wbf, deform_b, nullptr, out1bf, COUT, (long)KDIM);

  // 5) linear GEMM + bias + softshrink -> fp32 output
  gemm_bf16_wmma<2><<<dim3(BB / 64, COUT / 128), 128, 0, stream>>>(
      out1bf, linwbf, lin_b, out, nullptr, COUT, (long)COUT);
}
